// Attention_11527692222464
// MI455X (gfx1250) — matrix operand order repaired and checked by reading, not compiled, not run on hardware
//
#include <hip/hip_runtime.h>
#include <hip/hip_bf16.h>

typedef __attribute__((ext_vector_type(16))) __bf16 v16bf;
typedef __attribute__((ext_vector_type(8)))  float  v8f;

static constexpr int B = 4, N = 2048, I = 256, O = 128, H = 8;
static constexpr float NEG = -1.0e10f;

// ---------------------------------------------------------------------------
// K1: Wh[b,h,n,o] = sum_i h[b,n,i] * conv_w[h,o,i] + conv_b[h,o]
// Block = 256 threads = 8 waves. Block handles 16 rows x all 128 cols.
// Wave w computes the 16x16 tile at o0 = 16*w via v_wmma_f32_16x16x32_bf16.
// ---------------------------------------------------------------------------
__global__ __launch_bounds__(256) void gemm_wh_kernel(
    const float* __restrict__ hin, const float* __restrict__ conv_w,
    const float* __restrict__ conv_b, float* __restrict__ Wh) {
  const int m0 = blockIdx.x * 16;            // row tile
  const int bh = blockIdx.y;                 // b*H + h
  const int b  = bh / H;
  const int hd = bh % H;

  const int lane  = threadIdx.x & 31;
  const int wave  = threadIdx.x >> 5;
  const int lhalf = lane >> 4;               // 0: lanes 0-15, 1: lanes 16-31
  const int l15   = lane & 15;
  const int o0    = wave * 16;
  const int ocol  = o0 + l15;

  // A: row = m0 + (lane&15). lanes 0-15 hold K {0..7, 16..23}; lanes 16-31 hold {8..15, 24..31}
  const float* Arow = hin + ((size_t)b * N + (m0 + l15)) * I;
  // B[k][o] = conv_w[h, o, k]. lane column o = o0+l15; lanes 0-15 hold K {0..7, 16..23}; lanes 16-31 hold {8..15, 24..31}
  const float* Brow = conv_w + ((size_t)hd * O + ocol) * I;

  // Accumulator seeded with conv_b[h, o] (broadcast over the 8 row-VGPRs)
  const float bias = conv_b[hd * O + ocol];
  v8f acc;
#pragma unroll
  for (int r = 0; r < 8; ++r) acc[r] = bias;

#pragma unroll
  for (int k0 = 0; k0 < I; k0 += 32) {
    const int abase = k0 + (lhalf ? 8 : 0);
    const float4 a0 = *(const float4*)(Arow + abase);
    const float4 a1 = *(const float4*)(Arow + abase + 4);
    const float4 a2 = *(const float4*)(Arow + abase + 16);
    const float4 a3 = *(const float4*)(Arow + abase + 20);
    v16bf amat;
    amat[0]=(__bf16)a0.x;  amat[1]=(__bf16)a0.y;  amat[2]=(__bf16)a0.z;  amat[3]=(__bf16)a0.w;
    amat[4]=(__bf16)a1.x;  amat[5]=(__bf16)a1.y;  amat[6]=(__bf16)a1.z;  amat[7]=(__bf16)a1.w;
    amat[8]=(__bf16)a2.x;  amat[9]=(__bf16)a2.y;  amat[10]=(__bf16)a2.z; amat[11]=(__bf16)a2.w;
    amat[12]=(__bf16)a3.x; amat[13]=(__bf16)a3.y; amat[14]=(__bf16)a3.z; amat[15]=(__bf16)a3.w;

    const int bbase = k0 + (lhalf ? 8 : 0);
    const float4 b0 = *(const float4*)(Brow + bbase);
    const float4 b1 = *(const float4*)(Brow + bbase + 4);
    const float4 b2 = *(const float4*)(Brow + bbase + 16);
    const float4 b3 = *(const float4*)(Brow + bbase + 20);
    v16bf bmat;
    bmat[0]=(__bf16)b0.x;  bmat[1]=(__bf16)b0.y;  bmat[2]=(__bf16)b0.z;  bmat[3]=(__bf16)b0.w;
    bmat[4]=(__bf16)b1.x;  bmat[5]=(__bf16)b1.y;  bmat[6]=(__bf16)b1.z;  bmat[7]=(__bf16)b1.w;
    bmat[8]=(__bf16)b2.x;  bmat[9]=(__bf16)b2.y;  bmat[10]=(__bf16)b2.z; bmat[11]=(__bf16)b2.w;
    bmat[12]=(__bf16)b3.x; bmat[13]=(__bf16)b3.y; bmat[14]=(__bf16)b3.z; bmat[15]=(__bf16)b3.w;

    acc = __builtin_amdgcn_wmma_f32_16x16x32_bf16(
        /*neg_a=*/false, amat, /*neg_b=*/false, bmat,
        /*c_mod=*/(short)0, acc, /*reuse_a=*/false, /*reuse_b=*/false);
  }

  // D layout: VGPR r -> row M = r (lanes 0-15) or 8+r (lanes 16-31), col N = lane&15
  float* Wout = Wh + (((size_t)bh) * N + m0 + (lhalf ? 8 : 0)) * O + ocol;
#pragma unroll
  for (int r = 0; r < 8; ++r) Wout[(size_t)r * O] = acc[r];
}

// ---------------------------------------------------------------------------
// K2: r[b,h,n] = Wh[b,h,n,:].a1[h] + Wh1_bias[h,n] + Wh2_bias[h,n]
//     c[b,h,n] = Wh[b,h,n,:].a2[h]
// One 128-thread block per (b,h,n).
// ---------------------------------------------------------------------------
__global__ __launch_bounds__(128) void rowdot_kernel(
    const float* __restrict__ Wh, const float* __restrict__ a,
    const float* __restrict__ wh1b, const float* __restrict__ wh2b,
    float* __restrict__ rbuf, float* __restrict__ cbuf) {
  const size_t bhn = blockIdx.x;
  const int n  = (int)(bhn % N);
  const int hd = (int)((bhn / N) % H);
  const int tid = threadIdx.x;

  const float wh = Wh[bhn * O + tid];
  float p1 = wh * a[hd * 2 * O + tid];
  float p2 = wh * a[hd * 2 * O + O + tid];
#pragma unroll
  for (int off = 16; off > 0; off >>= 1) {
    p1 += __shfl_down(p1, off);
    p2 += __shfl_down(p2, off);
  }
  __shared__ float s1[4], s2[4];
  if ((tid & 31) == 0) { s1[tid >> 5] = p1; s2[tid >> 5] = p2; }
  __syncthreads();
  if (tid == 0) {
    const float t1 = s1[0] + s1[1] + s1[2] + s1[3];
    const float t2 = s2[0] + s2[1] + s2[2] + s2[3];
    rbuf[bhn] = t1 + wh1b[(size_t)hd * N + n] + wh2b[(size_t)hd * N + n];
    cbuf[bhn] = t2;
  }
}

// ---------------------------------------------------------------------------
// K3: online softmax over row j; only diagonal attention value is kept.
// score[j] = adj[b,n,j]>=0.5 ? leaky_relu(r + c[j]) + a_bias[h,n,j] : NEG
// One 256-thread block per (b,h,n).
// ---------------------------------------------------------------------------
__global__ __launch_bounds__(256) void diag_softmax_kernel(
    const float* __restrict__ adj, const float* __restrict__ a_bias,
    const float* __restrict__ rbuf, const float* __restrict__ cbuf,
    float* __restrict__ diag) {
  const size_t bhn = blockIdx.x;
  const int n  = (int)(bhn % N);
  const int hd = (int)((bhn / N) % H);
  const int b  = (int)(bhn / ((size_t)N * H));
  const int tid = threadIdx.x;

  const float rv = rbuf[bhn];
  const float* adjrow = adj + ((size_t)b * N + n) * N;
  const float* abrow  = a_bias + ((size_t)hd * N + n) * N;
  const float* crow   = cbuf + ((size_t)b * H + hd) * N;

  float m = -3.0e38f, s = 0.0f, dv = 0.0f;
#pragma unroll
  for (int j = tid; j < N; j += 256) {
    float sc;
    if (adjrow[j] >= 0.5f) {
      float x = rv + crow[j];
      x = (x >= 0.0f) ? x : 0.2f * x;
      sc = x + abrow[j];
    } else {
      sc = NEG;
    }
    const float nm = fmaxf(m, sc);
    s = s * __expf(m - nm) + __expf(sc - nm);
    m = nm;
    if (j == n) dv = sc;
  }
  // wave reduce (max, sum) pairs
#pragma unroll
  for (int off = 16; off > 0; off >>= 1) {
    const float om = __shfl_down(m, off);
    const float os = __shfl_down(s, off);
    const float nm = fmaxf(m, om);
    s = s * __expf(m - nm) + os * __expf(om - nm);
    m = nm;
  }
  __shared__ float lm[8], ls[8], ldv;
  if (tid == (n & 255)) ldv = dv;
  if ((tid & 31) == 0) { lm[tid >> 5] = m; ls[tid >> 5] = s; }
  __syncthreads();
  if (tid == 0) {
    float M = lm[0], S = ls[0];
#pragma unroll
    for (int w = 1; w < 8; ++w) {
      const float nm = fmaxf(M, lm[w]);
      S = S * __expf(M - nm) + ls[w] * __expf(lm[w] - nm);
      M = nm;
    }
    diag[bhn] = __expf(ldv - M) / S;
  }
}

// ---------------------------------------------------------------------------
// K4: out[b,n,h*O+o] = elu(diag[b,h,n] * Wh[b,h,n,o] + attention_bias[h,n,o])
// ---------------------------------------------------------------------------
__global__ __launch_bounds__(256) void elu_out_kernel(
    const float* __restrict__ Wh, const float* __restrict__ diag,
    const float* __restrict__ att_bias, float* __restrict__ out) {
  const size_t t = (size_t)blockIdx.x * 256 + threadIdx.x;
  const int o  = (int)(t % O);
  const int hd = (int)((t / O) % H);
  const int n  = (int)((t / (O * H)) % N);
  const int b  = (int)(t / ((size_t)O * H * N));

  const float wh = Wh[(((size_t)b * H + hd) * N + n) * O + o];
  const float d  = diag[((size_t)b * H + hd) * N + n];
  const float ab = att_bias[((size_t)hd * N + n) * O + o];
  const float v  = d * wh + ab;
  out[t] = (v > 0.0f) ? v : (__expf(v) - 1.0f);
}

extern "C" void kernel_launch(void* const* d_in, const int* in_sizes, int n_in,
                              void* d_out, int out_size, void* d_ws, size_t ws_size,
                              hipStream_t stream) {
  const float* hin      = (const float*)d_in[0];  // (B,N,I)
  const float* adj      = (const float*)d_in[1];  // (B,N,N)
  const float* conv_w   = (const float*)d_in[2];  // (H,O,I)
  const float* conv_b   = (const float*)d_in[3];  // (H,O)
  const float* a        = (const float*)d_in[4];  // (H,2O)
  const float* wh1b     = (const float*)d_in[5];  // (H,N,1)
  const float* wh2b     = (const float*)d_in[6];  // (H,N,1)
  const float* a_bias   = (const float*)d_in[7];  // (H,N,N)
  const float* att_bias = (const float*)d_in[8];  // (H,N,O)
  float* out = (float*)d_out;                     // (B,N,H*O)

  // workspace layout
  char* ws = (char*)d_ws;
  float* Wh   = (float*)ws;                                      // B*H*N*O floats (33.5 MB)
  float* rbuf = (float*)(ws + (size_t)B * H * N * O * 4);        // B*H*N
  float* cbuf = rbuf + (size_t)B * H * N;
  float* diag = cbuf + (size_t)B * H * N;

  // K1: GEMM with WMMA bf16
  dim3 g1(N / 16, B * H);
  gemm_wh_kernel<<<g1, 256, 0, stream>>>(hin, conv_w, conv_b, Wh);

  // K2: per-row dots
  rowdot_kernel<<<(uint32_t)((size_t)B * H * N), 128, 0, stream>>>(
      Wh, a, wh1b, wh2b, rbuf, cbuf);

  // K3: masked row softmax, diagonal only
  diag_softmax_kernel<<<(uint32_t)((size_t)B * H * N), 256, 0, stream>>>(
      adj, a_bias, rbuf, cbuf, diag);

  // K4: elu epilogue + transpose-on-store
  const size_t total = (size_t)B * N * H * O;
  elu_out_kernel<<<(uint32_t)(total / 256), 256, 0, stream>>>(
      Wh, diag, att_bias, out);
}
